// DeepConvWeigthNet_32856499814948
// MI455X (gfx1250) — matrix operand order repaired and checked by reading, not compiled, not run on hardware
//
#include <hip/hip_runtime.h>
#include <hip/hip_bf16.h>

typedef __attribute__((ext_vector_type(16))) __bf16 v16bf;
typedef __attribute__((ext_vector_type(8)))  float  v8f;

#define BATCH 8
#define HH 512
#define WW 512
#define NPIX (HH * WW)           // 262144
#define TOTPIX (BATCH * NPIX)    // 2097152

// ---------------------------------------------------------------------------
// Async global->LDS 16B copy (CDNA5 GLOBAL_LOAD_ASYNC_TO_LDS_B128 path),
// guarded so compile succeeds on any toolchain. #pragma message probes report
// which path was taken via compiler stderr.
// ---------------------------------------------------------------------------
#if defined(__AMDGCN__)
#if __has_builtin(__builtin_amdgcn_load_to_lds)
#pragma message("CDNA5 probe: using __builtin_amdgcn_load_to_lds (async path)")
#elif __has_builtin(__builtin_amdgcn_global_load_lds)
#pragma message("CDNA5 probe: using __builtin_amdgcn_global_load_lds")
#else
#pragma message("CDNA5 probe: NO async-to-LDS builtin; sync fallback in use")
#endif
#endif

__device__ __forceinline__ void async_copy16(const void* gsrc, void* ldst) {
#if defined(__AMDGCN__) && __has_builtin(__builtin_amdgcn_load_to_lds)
    __builtin_amdgcn_load_to_lds(
        (const __attribute__((address_space(1))) void*)gsrc,
        (__attribute__((address_space(3))) void*)(unsigned int)(unsigned long long)ldst,
        16, 0, 0);
#elif defined(__AMDGCN__) && __has_builtin(__builtin_amdgcn_global_load_lds)
    __builtin_amdgcn_global_load_lds(
        (const __attribute__((address_space(1))) unsigned int*)gsrc,
        (__attribute__((address_space(3))) unsigned int*)(unsigned int)(unsigned long long)ldst,
        16, 0, 0);
#else
    *(int4*)ldst = *(const int4*)gsrc;
#endif
}

__device__ __forceinline__ void wait_async_lds() {
#if defined(__AMDGCN__)
#if __has_builtin(__builtin_amdgcn_s_wait_asynccnt)
    __builtin_amdgcn_s_wait_asynccnt(0);
#else
    asm volatile("s_wait_asynccnt 0x0" ::: "memory");
#endif
#endif
}

// ---------------------------------------------------------------------------
// Weight packing: OIHW fp32 -> [chunk t][n][kl] bf16, t = tap*(CIN/32)+cc,
// kl = channel within 32-wide K chunk. Matches WMMA B-fragment loads.
// ---------------------------------------------------------------------------
__global__ __launch_bounds__(256) void pack_weights(const float* __restrict__ w,
                                                    __bf16* __restrict__ out,
                                                    int COUT, int CIN) {
    int i = blockIdx.x * 256 + threadIdx.x;
    int total = 9 * CIN * COUT;
    if (i >= total) return;
    int kl = i & 31;
    int n  = (i >> 5) % COUT;
    int t  = i / (32 * COUT);
    int CC = CIN >> 5;
    int tap = t / CC, cc = t % CC;
    int c = cc * 32 + kl;
    out[i] = (__bf16)w[(n * CIN + c) * 9 + tap];
}

// Pack the three 4x32x3x3 head convs into one N=16 (12 real, 4 zero) buffer.
__global__ __launch_bounds__(256) void pack_head_weights(
    const float* __restrict__ w1, const float* __restrict__ w2, const float* __restrict__ w3,
    const float* __restrict__ b1, const float* __restrict__ b2, const float* __restrict__ b3,
    __bf16* __restrict__ wOut, float* __restrict__ bOut) {
    int i = blockIdx.x * 256 + threadIdx.x;
    if (i < 9 * 16 * 32) {
        int kl = i & 31;
        int n  = (i >> 5) & 15;
        int tap = i / (16 * 32);
        float v = 0.f;
        if (n < 12) {
            const float* w = (n < 4) ? w1 : (n < 8) ? w2 : w3;
            int nn = n & 3;
            v = w[(nn * 32 + kl) * 9 + tap];
        }
        wOut[(tap * 16 + n) * 32 + kl] = (__bf16)v;
    }
    if (i < 16) {
        float bv = 0.f;
        if (i < 12) {
            const float* b = (i < 4) ? b1 : (i < 8) ? b2 : b3;
            bv = b[i & 3];
        }
        bOut[i] = bv;
    }
}

// ---------------------------------------------------------------------------
// conv1: NCHW fp32 x (Cin=3) -> NHWC bf16 body1 (Cout=32), 3x3 pad 1, PReLU.
// K=27: too small for WMMA; direct VALU conv, weights staged in LDS.
// ---------------------------------------------------------------------------
__global__ __launch_bounds__(256) void conv1_prelu(
    const float* __restrict__ x, const float* __restrict__ w,
    const float* __restrict__ bias, const float* __restrict__ alpha,
    __bf16* __restrict__ out) {
    __shared__ float wsh[32 * 27];
    for (int i = threadIdx.x; i < 32 * 27; i += 256) wsh[i] = w[i];
    __syncthreads();

    int pix = blockIdx.x * 256 + threadIdx.x;       // [0, TOTPIX)
    int b = pix >> 18;
    int p = pix & (NPIX - 1);
    int y = p >> 9, xx = p & 511;

    float patch[27];
#pragma unroll
    for (int c = 0; c < 3; ++c)
#pragma unroll
        for (int t = 0; t < 9; ++t) {
            int gy = y + t / 3 - 1, gx = xx + t % 3 - 1;
            patch[c * 9 + t] = ((unsigned)gy < 512u && (unsigned)gx < 512u)
                ? x[(((size_t)(b * 3 + c)) << 18) + ((size_t)gy << 9) + gx]
                : 0.f;
        }
    float aval = alpha[0];
#pragma unroll 4
    for (int oc = 0; oc < 32; ++oc) {
        float s = bias[oc];
#pragma unroll
        for (int k = 0; k < 27; ++k) s = fmaf(wsh[oc * 27 + k], patch[k], s);
        s = s > 0.f ? s : aval * s;
        out[((size_t)pix << 5) + oc] = (__bf16)s;
    }
}

// ---------------------------------------------------------------------------
// Implicit-GEMM 3x3 conv via WMMA bf16. NHWC bf16 in -> NHWC out.
// Block: 256 thr = 8 waves; tile = 128 px (x) * ROWS=4 rows (y) * COUT.
// The (ROWS+2)-row LDS halo tile cuts input read amplification from 3x to
// 1.5x (critical: body2 = 268MB exceeds the 192MB L2, so this is HBM BW),
// and amortizes the per-block LDS weight staging 4x. Wave computes 16 px x
// COUT per row-iteration; K = CIN*9 in chunks of 32. B fragments come from
// LDS (8-wave reuse); halo fill uses async global->LDS copies when declared.
// ---------------------------------------------------------------------------
template <int CIN, int COUT, bool DO_PRELU, bool F32OUT>
__global__ __launch_bounds__(256) void conv3x3_wmma(
    const __bf16* __restrict__ in, const __bf16* __restrict__ wB,
    const float* __restrict__ bias, const float* __restrict__ alpha,
    __bf16* __restrict__ out_bf, float* __restrict__ out_f) {
    constexpr int WT = 128;
    constexpr int ROWS = 4;
    constexpr int CC = CIN / 32;
    constexpr int NT = COUT / 16;
    constexpr int NCHUNK = 9 * CC;
    constexpr int PXCH = CIN / 8;            // 16B chunks per pixel
    __shared__ __bf16 lds_in[(ROWS + 2) * (WT + 2) * CIN];
    __shared__ __bf16 lds_w[NCHUNK * COUT * 32];

    const int tid  = threadIdx.x;
    const int lane = tid & 31;
    const int wave = tid >> 5;
    const int x0 = blockIdx.x * WT;
    const int y0 = blockIdx.y * ROWS;
    const int b  = blockIdx.z;

    // Stage the packed weights once per block (coalesced, L2-resident source).
    for (int i = tid; i < NCHUNK * COUT * 4; i += 256)
        ((int4*)lds_w)[i] = ((const int4*)wB)[i];

    // Halo tile: rows y0-1..y0+ROWS, x0-1..x0+128, zero padded, async to LDS.
    for (int i = tid; i < (ROWS + 2) * (WT + 2) * PXCH; i += 256) {
        int e = i / PXCH, j = i % PXCH;
        int r = e / (WT + 2), xl = e % (WT + 2);
        int gy = y0 + r - 1, gx = x0 + xl - 1;
        void* dst = (char*)lds_in + (size_t)i * 16;
        if ((unsigned)gy < 512u && (unsigned)gx < 512u) {
            async_copy16(&in[(((size_t)b * HH + gy) * WW + gx) * CIN + j * 8], dst);
        } else {
            int4 z; z.x = z.y = z.z = z.w = 0;
            *(int4*)dst = z;
        }
    }
    wait_async_lds();
    __syncthreads();

    const int m    = lane & 15;   // A-matrix row / C,B-matrix N column owner
    const int hi   = lane >> 4;   // K-half selector per ISA bf16 A/B layouts
    const int px0  = wave * 16;
    const int ncol = m;
    const float aval = DO_PRELU ? alpha[0] : 0.f;

    for (int rr = 0; rr < ROWS; ++rr) {
        v8f acc[NT];
#pragma unroll
        for (int nt = 0; nt < NT; ++nt) {
            float bv = bias[nt * 16 + ncol];
#pragma unroll
            for (int v = 0; v < 8; ++v) acc[nt][v] = bv;
        }

#pragma unroll
        for (int tap = 0; tap < 9; ++tap) {
            const int ky = tap / 3, kx = tap % 3;
#pragma unroll
            for (int cc = 0; cc < CC; ++cc) {
                // A fragment: lanes 0-15 K{0..7,16..23}, 16-31 K{8..15,24..31}.
                const __bf16* pix =
                    &lds_in[((size_t)((rr + ky) * (WT + 2)) + (px0 + m + kx)) * CIN
                            + cc * 32];
                v16bf afrag;
                ((int4*)&afrag)[0] = *(const int4*)(pix + hi * 8);
                ((int4*)&afrag)[1] = *(const int4*)(pix + 16 + hi * 8);

                const int t = tap * CC + cc;
#pragma unroll
                for (int nt = 0; nt < NT; ++nt) {
                    // B fragment: lane owns column n, K{0..7,16..23} / K{8..15,24..31} (LDS).
                    const __bf16* wp =
                        &lds_w[((size_t)(t * COUT) + nt * 16 + ncol) * 32 + hi * 8];
                    v16bf bfrag;
                    ((int4*)&bfrag)[0] = *(const int4*)(wp);
                    ((int4*)&bfrag)[1] = *(const int4*)(wp + 16);
                    acc[nt] = __builtin_amdgcn_wmma_f32_16x16x32_bf16(
                        false, afrag, false, bfrag, (short)0, acc[nt], false, false);
                }
            }
        }

        const int y = y0 + rr;
#pragma unroll
        for (int nt = 0; nt < NT; ++nt) {
#pragma unroll
            for (int v = 0; v < 8; ++v) {
                int mm = v + hi * 8;             // C layout: M = v (+8 hi half)
                int gx = x0 + px0 + mm;
                float f = acc[nt][v];
                if (DO_PRELU) f = f > 0.f ? f : aval * f;
                size_t o = (((size_t)b * HH + y) * WW + gx) * COUT + nt * 16 + ncol;
                if (F32OUT) out_f[o] = f; else out_bf[o] = (__bf16)f;
            }
        }
    }
}

// ---------------------------------------------------------------------------
// Global average pool over HxW for the 12 head channels (NHWC16 f32 input).
// ---------------------------------------------------------------------------
__global__ __launch_bounds__(256) void gap12(const float* __restrict__ h,
                                             float* __restrict__ g) {
    int b = blockIdx.x / 12, ch = blockIdx.x % 12;
    const float* base = h + ((size_t)b * NPIX) * 16 + ch;
    float s = 0.f;
    for (int i = threadIdx.x; i < NPIX; i += 256) s += base[(size_t)i * 16];
    __shared__ float red[256];
    red[threadIdx.x] = s;
    __syncthreads();
    for (int off = 128; off > 0; off >>= 1) {
        if (threadIdx.x < off) red[threadIdx.x] += red[threadIdx.x + off];
        __syncthreads();
    }
    if (threadIdx.x == 0) g[blockIdx.x] = red[0] * (1.f / (float)NPIX);
}

// Channel attention: s = sigmoid(c2 @ relu(c1 @ g)) per (batch, head).
__global__ void ca_kernel(const float* __restrict__ g,
                          const float* __restrict__ c11, const float* __restrict__ c12,
                          const float* __restrict__ c21, const float* __restrict__ c22,
                          const float* __restrict__ c31, const float* __restrict__ c32,
                          float* __restrict__ s) {
    int t = threadIdx.x;
    if (t >= 96) return;
    int b = t / 12, ch = t % 12, head = ch >> 2, i = ch & 3;
    const float* c1 = head == 0 ? c11 : head == 1 ? c21 : c31;
    const float* c2 = head == 0 ? c12 : head == 1 ? c22 : c32;
    const float* gg = g + b * 12 + head * 4;
    float tv[4];
#pragma unroll
    for (int j = 0; j < 4; ++j) {
        float a = 0.f;
#pragma unroll
        for (int k = 0; k < 4; ++k) a = fmaf(c1[j * 4 + k], gg[k], a);
        tv[j] = fmaxf(a, 0.f);
    }
    float v = 0.f;
#pragma unroll
    for (int j = 0; j < 4; ++j) v = fmaf(c2[i * 4 + j], tv[j], v);
    s[b * 12 + ch] = 1.f / (1.f + __expf(-v));
}

// ---------------------------------------------------------------------------
// Separable box blurs: one horizontal pass computes all three radii (2/7/12)
// sharing the 25-tap window, then a vertical pass. NCHW f32 throughout.
// ---------------------------------------------------------------------------
__global__ __launch_bounds__(256) void blur3h(const float* __restrict__ in,
                                              float* __restrict__ o2,
                                              float* __restrict__ o3,
                                              float* __restrict__ o4) {
    int idx = blockIdx.x * 256 + threadIdx.x;     // 24 * 512 * 512
    int bc = idx >> 18;
    int p  = idx & (NPIX - 1);
    int y = p >> 9, x = p & 511;
    const float* row = in + ((size_t)bc << 18) + ((size_t)y << 9);
    float s5 = 0.f, s15 = 0.f, s25 = 0.f;
#pragma unroll
    for (int dx = -12; dx <= 12; ++dx) {
        int xx = x + dx;
        float v = ((unsigned)xx < 512u) ? row[xx] : 0.f;
        s25 += v;
        if (dx >= -7 && dx <= 7) s15 += v;
        if (dx >= -2 && dx <= 2) s5 += v;
    }
    size_t o = ((size_t)bc << 18) + p;
    o2[o] = s5 * 0.2f;
    o3[o] = s15 * (1.f / 15.f);
    o4[o] = s25 * (1.f / 25.f);
}

__global__ __launch_bounds__(256) void blur3v(const float* __restrict__ t2,
                                              const float* __restrict__ t3,
                                              const float* __restrict__ t4,
                                              float* __restrict__ m2,
                                              float* __restrict__ m3,
                                              float* __restrict__ m4) {
    int idx = blockIdx.x * 256 + threadIdx.x;
    int bc = idx >> 18;
    int p  = idx & (NPIX - 1);
    int y = p >> 9, x = p & 511;
    size_t base = ((size_t)bc << 18) + x;
    float s5 = 0.f, s15 = 0.f, s25 = 0.f;
#pragma unroll
    for (int dy = -12; dy <= 12; ++dy) {
        int yy = y + dy;
        if ((unsigned)yy < 512u) {
            size_t o = base + ((size_t)yy << 9);
            float v4 = t4[o];
            s25 += v4;
            if (dy >= -7 && dy <= 7) s15 += t3[o];
            if (dy >= -2 && dy <= 2) s5 += t2[o];
        }
    }
    size_t o = ((size_t)bc << 18) + p;
    m2[o] = s5 * 0.2f;
    m3[o] = s15 * (1.f / 15.f);
    m4[o] = s25 * (1.f / 25.f);
}

// Fuse: softmax over the head's 4 channels of (h*s), then blend the 4 images.
__global__ __launch_bounds__(256) void fuse_k(const float* __restrict__ cur,
                                              const float* __restrict__ m2,
                                              const float* __restrict__ m3,
                                              const float* __restrict__ m4,
                                              const float* __restrict__ h_all,
                                              const float* __restrict__ s,
                                              int head, float* __restrict__ out) {
    int pix = blockIdx.x * 256 + threadIdx.x;     // TOTPIX
    int b = pix >> 18;
    int p = pix & (NPIX - 1);
    float e[4];
    float mx = -1e30f;
#pragma unroll
    for (int i = 0; i < 4; ++i) {
        e[i] = h_all[(size_t)pix * 16 + head * 4 + i] * s[b * 12 + head * 4 + i];
        mx = fmaxf(mx, e[i]);
    }
    float sum = 0.f;
#pragma unroll
    for (int i = 0; i < 4; ++i) { e[i] = __expf(e[i] - mx); sum += e[i]; }
    float r = 1.f / sum;
#pragma unroll
    for (int i = 0; i < 4; ++i) e[i] *= r;
#pragma unroll
    for (int c = 0; c < 3; ++c) {
        size_t o = ((size_t)(b * 3 + c) << 18) + p;
        out[o] = e[0] * cur[o] + e[1] * m2[o] + e[2] * m3[o] + e[3] * m4[o];
    }
}

// ---------------------------------------------------------------------------
// Host-side orchestration.
// ---------------------------------------------------------------------------
extern "C" void kernel_launch(void* const* d_in, const int* in_sizes, int n_in,
                              void* d_out, int out_size, void* d_ws, size_t ws_size,
                              hipStream_t stream) {
    const float* x   = (const float*)d_in[0];
    const float* bw1 = (const float*)d_in[1];
    const float* bb1 = (const float*)d_in[2];
    const float* a1  = (const float*)d_in[3];
    const float* bw2 = (const float*)d_in[4];
    const float* bb2 = (const float*)d_in[5];
    const float* a2  = (const float*)d_in[6];
    const float* bw3 = (const float*)d_in[7];
    const float* bb3 = (const float*)d_in[8];
    const float* a3  = (const float*)d_in[9];
    const float* h1w = (const float*)d_in[10];
    const float* h1b = (const float*)d_in[11];
    const float* h1c1 = (const float*)d_in[12];
    const float* h1c2 = (const float*)d_in[13];
    const float* h2w = (const float*)d_in[14];
    const float* h2b = (const float*)d_in[15];
    const float* h2c1 = (const float*)d_in[16];
    const float* h2c2 = (const float*)d_in[17];
    const float* h3w = (const float*)d_in[18];
    const float* h3b = (const float*)d_in[19];
    const float* h3c1 = (const float*)d_in[20];
    const float* h3c2 = (const float*)d_in[21];

    char* ws = (char*)d_ws;
    const size_t SZ_B1 = (size_t)TOTPIX * 32 * 2;   // 134,217,728
    const size_t SZ_B2 = (size_t)TOTPIX * 64 * 2;   // 268,435,456
    const size_t SZ_B3 = (size_t)TOTPIX * 32 * 2;
    const size_t SZ_H  = (size_t)TOTPIX * 16 * 4;
    const size_t SZ_IMG = (size_t)TOTPIX * 3 * 4;   // 25,165,824

    size_t off = 0;
    __bf16* body1 = (__bf16*)(ws + off); off += SZ_B1;
    size_t off_b2 = off;
    __bf16* body2 = (__bf16*)(ws + off); off += SZ_B2;
    __bf16* body3 = (__bf16*)(ws + off); off += SZ_B3;
    float*  h_all = (float*)(ws + off);  off += SZ_H;
    __bf16* wB2 = (__bf16*)(ws + off); off += 9 * 32 * 64 * 2;
    __bf16* wB3 = (__bf16*)(ws + off); off += 9 * 64 * 32 * 2;
    __bf16* wBh = (__bf16*)(ws + off); off += 9 * 16 * 32 * 2;
    float* hb = (float*)(ws + off); off += 256;
    float* g  = (float*)(ws + off); off += 512;
    float* sA = (float*)(ws + off); off += 512;

    // Blur temporaries alias body1 (dead after conv2 consumes it);
    // m-buffers + cur ping-pong alias body2 (dead after conv3).
    float* tH2 = (float*)(ws + 0);
    float* tH3 = (float*)(ws + SZ_IMG);
    float* tH4 = (float*)(ws + 2 * SZ_IMG);
    float* m2  = (float*)(ws + off_b2);
    float* m3  = (float*)(ws + off_b2 + SZ_IMG);
    float* m4  = (float*)(ws + off_b2 + 2 * SZ_IMG);
    float* curA = (float*)(ws + off_b2 + 3 * SZ_IMG);
    float* curB = (float*)(ws + off_b2 + 4 * SZ_IMG);

    // Weight packing (tiny).
    pack_weights<<<(9 * 32 * 64 + 255) / 256, 256, 0, stream>>>(bw2, wB2, 64, 32);
    pack_weights<<<(9 * 64 * 32 + 255) / 256, 256, 0, stream>>>(bw3, wB3, 32, 64);
    pack_head_weights<<<(9 * 16 * 32 + 255) / 256, 256, 0, stream>>>(
        h1w, h2w, h3w, h1b, h2b, h3b, wBh, hb);

    // Body.
    conv1_prelu<<<TOTPIX / 256, 256, 0, stream>>>(x, bw1, bb1, a1, body1);
    dim3 cgrid(WW / 128, HH / 4, BATCH);
    conv3x3_wmma<32, 64, true, false><<<cgrid, 256, 0, stream>>>(
        body1, wB2, bb2, a2, body2, nullptr);
    conv3x3_wmma<64, 32, true, false><<<cgrid, 256, 0, stream>>>(
        body2, wB3, bb3, a3, body3, nullptr);

    // Heads (all three fused into one N=16 WMMA conv, f32 out).
    conv3x3_wmma<32, 16, false, true><<<cgrid, 256, 0, stream>>>(
        body3, wBh, hb, nullptr, nullptr, h_all);
    gap12<<<96, 256, 0, stream>>>(h_all, g);
    ca_kernel<<<1, 128, 0, stream>>>(g, h1c1, h1c2, h2c1, h2c2, h3c1, h3c2, sA);

    const int BLUR_B = (24 * NPIX) / 256;
    const int FUSE_B = TOTPIX / 256;

    // Stage 1: fuse(x, h1) -> curA
    blur3h<<<BLUR_B, 256, 0, stream>>>(x, tH2, tH3, tH4);
    blur3v<<<BLUR_B, 256, 0, stream>>>(tH2, tH3, tH4, m2, m3, m4);
    fuse_k<<<FUSE_B, 256, 0, stream>>>(x, m2, m3, m4, h_all, sA, 0, curA);

    // Stage 2: fuse(curA, h2) -> curB
    blur3h<<<BLUR_B, 256, 0, stream>>>(curA, tH2, tH3, tH4);
    blur3v<<<BLUR_B, 256, 0, stream>>>(tH2, tH3, tH4, m2, m3, m4);
    fuse_k<<<FUSE_B, 256, 0, stream>>>(curA, m2, m3, m4, h_all, sA, 1, curB);

    // Stage 3: fuse(curB, h3) -> d_out
    blur3h<<<BLUR_B, 256, 0, stream>>>(curB, tH2, tH3, tH4);
    blur3v<<<BLUR_B, 256, 0, stream>>>(tH2, tH3, tH4, m2, m3, m4);
    fuse_k<<<FUSE_B, 256, 0, stream>>>(curB, m2, m3, m4, h_all, sA, 2, (float*)d_out);
}
